// GlobalCorrelationMatcher_20040317403411
// MI455X (gfx1250) — matrix operand order repaired and checked by reading, not compiled, not run on hardware
//
#include <hip/hip_runtime.h>

// ---------------------------------------------------------------------------
// GlobalCorrelationMatcher for MI455X (gfx1250, wave32, WMMA)
//
//   K1 proj_norm : per-pixel 1x1-conv projections q,k (+bias), L2-normalize
//                  over P=128, store as f16 [b][pixel][128] in workspace.
//   K2 corr      : flash-style fused correlation + softmax + expectation.
//                  Workgroup = 8 waves, all covering the SAME 16 query rows.
//                  Wave w owns 72 CONTIGUOUS column tiles = 12 image rows
//                  (6 tiles per image row since W = 96 = 6*16), so:
//                    - y coordinate: one v_add per image row
//                    - x coordinate: 6 precomputed per-lane registers
//                    - B-tile addresses: pure immediate offsets
//                  Per tile: 4x v_wmma_f32_16x16x32_f16 (K=128), then
//                  e = exp2(dot*K - K), K = log2(e)/T folded into one fma
//                  (valid since cosine similarity <= 1 => no overflow).
//                  Fixed-max partials are additive -> waves combine via LDS.
//                  conf = exp((max-1)/T)/S falls out of the tracked max.
// ---------------------------------------------------------------------------

typedef _Float16 half8 __attribute__((ext_vector_type(8)));
typedef _Float16 v16h  __attribute__((ext_vector_type(16)));
typedef float    v8f   __attribute__((ext_vector_type(8)));

#define BB    2
#define CC    256
#define HH    96
#define WW    96
#define HWN   9216        // 96*96
#define PP    128
#define NW    8           // waves per workgroup in corr kernel
#define TPR   6           // column tiles per image row (96/16)
#define RPW   12          // image rows per wave (96/NW)
#define INVT  (1.0f/0.07f)
#define K2E   (INVT * 1.4426950408889634f)   // log2(e)/T
#define COORD_STEP (2.0f/95.0f)              // linspace(-1,1,96) step

union V16 { v16h v; half8 h[2]; };

// ---------------------------------------------------------------------------
// Kernel 1: projection + L2 normalize.  grid (HWN, B), block 128 (one per p).
// ---------------------------------------------------------------------------
__global__ __launch_bounds__(128)
void proj_norm_kernel(const float* __restrict__ fa, const float* __restrict__ fb,
                      const float* __restrict__ wq, const float* __restrict__ bq,
                      const float* __restrict__ wk, const float* __restrict__ bk,
                      _Float16* __restrict__ qa, _Float16* __restrict__ kb) {
    const int n = blockIdx.x;      // pixel
    const int b = blockIdx.y;      // batch
    const int p = threadIdx.x;     // output channel

    __shared__ float sfa[CC];
    __shared__ float sfb[CC];
    for (int c = p; c < CC; c += 128) {
        sfa[c] = fa[((size_t)(b * CC + c)) * HWN + n];
        sfb[c] = fb[((size_t)(b * CC + c)) * HWN + n];
    }
    __syncthreads();

    float q = bq[p];
    float k = bk[p];
    const float* wqr = wq + (size_t)p * CC;
    const float* wkr = wk + (size_t)p * CC;
#pragma unroll 8
    for (int c = 0; c < CC; ++c) {
        q = fmaf(sfa[c], wqr[c], q);
        k = fmaf(sfb[c], wkr[c], k);
    }

    __shared__ float rq[128];
    __shared__ float rk[128];
    rq[p] = q * q;
    rk[p] = k * k;
    __syncthreads();
    for (int s = 64; s > 0; s >>= 1) {
        if (p < s) { rq[p] += rq[p + s]; rk[p] += rk[p + s]; }
        __syncthreads();
    }
    const float nq = fmaxf(sqrtf(rq[0]), 1e-12f);
    const float nk = fmaxf(sqrtf(rk[0]), 1e-12f);

    const size_t base = ((size_t)b * HWN + n) * PP + p;
    qa[base] = (_Float16)(q / nq);
    kb[base] = (_Float16)(k / nk);
}

// ---------------------------------------------------------------------------
// Kernel 2: fused correlation + softmax + flow/conf.
// grid (HWN/16, B), block 256 = 8 waves; all waves share one 16-row tile.
// ---------------------------------------------------------------------------
__global__ __launch_bounds__(NW * 32)
void corr_softmax_kernel(const _Float16* __restrict__ qa,
                         const _Float16* __restrict__ kb,
                         float* __restrict__ out) {
    const int b    = blockIdx.y;
    const int n0   = blockIdx.x * 16;         // first query row of this block
    const int wave = threadIdx.x >> 5;
    const int lane = threadIdx.x & 31;
    const int nr   = lane & 15;               // index within half-group
    const int hi   = lane >> 4;               // half-group select

    // ---- A fragments: rows n0..n0+15, K=0..127 (4 chunks of 32) ----------
    // ISA 16-bit A 16x32 layout: lane L holds M=L&15,
    //   lo lanes: K = {0..7, 16..23}; hi lanes: K = {8..15, 24..31}.
    const _Float16* aptr = qa + ((size_t)b * HWN + n0 + nr) * PP;
    V16 av[4];
#pragma unroll
    for (int c4 = 0; c4 < 4; ++c4) {
        const int ko = 32 * c4 + hi * 8;
        av[c4].h[0] = *(const half8*)(aptr + ko);
        av[c4].h[1] = *(const half8*)(aptr + ko + 16);
    }

    // Per-lane accumulators: 8 rows (C/D VGPR r <-> row r + hi*8), 1 column.
    float s[8], cx[8], cy[8], vmax[8];
#pragma unroll
    for (int r = 0; r < 8; ++r) { s[r] = 0.f; cx[r] = 0.f; cy[r] = 0.f; vmax[r] = -2.0f; }

    // x coordinate of this lane's column within each of the 6 tiles of an
    // image row: column m = 16*j + nr  ->  6 per-lane constants.
    float xj[TPR];
#pragma unroll
    for (int j = 0; j < TPR; ++j)
        xj[j] = -1.f + (float)(16 * j + nr) * COORD_STEP;

    // Wave w owns image rows [w*RPW, (w+1)*RPW) of the key image.
    float y = -1.f + (float)(wave * RPW) * COORD_STEP;

    // Base pointer for this wave's first B tile (includes nr*PP).
    const _Float16* bptr =
        kb + ((size_t)b * HWN + (size_t)wave * RPW * WW + nr) * PP;

    for (int i = 0; i < RPW; ++i) {           // image rows of the key image
#pragma unroll
        for (int j = 0; j < TPR; ++j) {       // 6 tiles within the image row
            // ---- B fragment: cols 16j..16j+15 (dense 16-bit B layout):
            //      lane L holds N=L&15, K = {hi*8 + 0..7, 16 + hi*8 + 0..7}.
            const _Float16* tp = bptr + (size_t)j * 16 * PP;
            v8f acc = {};
#pragma unroll
            for (int c4 = 0; c4 < 4; ++c4) {
                const int ko = 32 * c4 + hi * 8;
                V16 bv;
                bv.h[0] = *(const half8*)(tp + ko);
                bv.h[1] = *(const half8*)(tp + ko + 16);
                acc = __builtin_amdgcn_wmma_f32_16x16x32_f16(
                    false, av[c4].v, false, bv.v, (short)0, acc, false, false);
            }
            // e = exp((dot-1)/T) = 2^(dot*K2E - K2E): one fma + one v_exp_f32.
            const float x = xj[j];
#pragma unroll
            for (int r = 0; r < 8; ++r) {
                const float v = acc[r];                   // dot product
                vmax[r] = fmaxf(vmax[r], v);
                const float e = __builtin_amdgcn_exp2f(fmaf(v, K2E, -K2E));
                s[r]  += e;
                cx[r]  = fmaf(e, x, cx[r]);
                cy[r]  = fmaf(e, y, cy[r]);
            }
        }
        bptr += (size_t)TPR * 16 * PP;        // advance one image row
        y    += COORD_STEP;
    }

    // ---- Reduce each row across its 16-lane half-group --------------------
#pragma unroll
    for (int r = 0; r < 8; ++r) {
#pragma unroll
        for (int m = 1; m <= 8; m <<= 1) {
            s[r]    += __shfl_xor(s[r],  m, 32);
            cx[r]   += __shfl_xor(cx[r], m, 32);
            cy[r]   += __shfl_xor(cy[r], m, 32);
            vmax[r]  = fmaxf(vmax[r], __shfl_xor(vmax[r], m, 32));
        }
    }

    // ---- Cross-wave combine through LDS (partials are additive) -----------
    __shared__ float ls [NW][16];
    __shared__ float lcx[NW][16];
    __shared__ float lcy[NW][16];
    __shared__ float lvm[NW][16];
    if (nr == 0) {
#pragma unroll
        for (int r = 0; r < 8; ++r) {
            const int row = r + hi * 8;
            ls [wave][row] = s[r];
            lcx[wave][row] = cx[r];
            lcy[wave][row] = cy[r];
            lvm[wave][row] = vmax[r];
        }
    }
    __syncthreads();

    // ---- Final combine + output: one thread per query row -----------------
    if (threadIdx.x < 16) {
        const int row = threadIdx.x;
        float S = 0.f, CX = 0.f, CY = 0.f, VM = -2.0f;
#pragma unroll
        for (int w = 0; w < NW; ++w) {
            S  += ls [w][row];
            CX += lcx[w][row];
            CY += lcy[w][row];
            VM  = fmaxf(VM, lvm[w][row]);
        }
        const int   n    = n0 + row;
        const float inv  = 1.0f / S;
        const float ex   = CX * inv;
        const float ey   = CY * inv;
        const float xn   = -1.f + (float)(n % WW) * COORD_STEP;
        const float yn   = -1.f + (float)(n / WW) * COORD_STEP;
        const float conf = __builtin_amdgcn_exp2f(fmaf(VM, K2E, -K2E)) * inv;
        out[((size_t)b * 2 + 0) * HWN + n] = (ex - xn) * (WW * 0.5f);
        out[((size_t)b * 2 + 1) * HWN + n] = (ey - yn) * (HH * 0.5f);
        out[(size_t)BB * 2 * HWN + (size_t)b * HWN + n] = conf;
    }
}

// ---------------------------------------------------------------------------
extern "C" void kernel_launch(void* const* d_in, const int* in_sizes, int n_in,
                              void* d_out, int out_size, void* d_ws, size_t ws_size,
                              hipStream_t stream) {
    const float* fa = (const float*)d_in[0];   // feat_a [2,256,96,96]
    const float* fb = (const float*)d_in[1];   // feat_b [2,256,96,96]
    const float* wq = (const float*)d_in[2];   // [128,256]
    const float* bq = (const float*)d_in[3];   // [128]
    const float* wk = (const float*)d_in[4];   // [128,256]
    const float* bk = (const float*)d_in[5];   // [128]
    float* out = (float*)d_out;                // flow [2,2,96,96] ++ conf [2,1,96,96]

    _Float16* qa = (_Float16*)d_ws;                          // 2*9216*128 f16
    _Float16* kb = qa + (size_t)BB * HWN * PP;               // 2*9216*128 f16
    (void)in_sizes; (void)n_in; (void)out_size; (void)ws_size;

    proj_norm_kernel<<<dim3(HWN, BB), 128, 0, stream>>>(fa, fb, wq, bq, wk, bk, qa, kb);
    corr_softmax_kernel<<<dim3(HWN / 16, BB), NW * 32, 0, stream>>>(qa, kb, out);
}
